// GatedMultiHeadAttention_14516989460913
// MI455X (gfx1250) — matrix operand order repaired and checked by reading, not compiled, not run on hardware
//
#include <hip/hip_runtime.h>
#include <hip/hip_bf16.h>

// ---------------------------------------------------------------------------
// GatedMultiHeadAttention for MI455X (gfx1250, wave32, WMMA).
// B=2, N=2048, L=D=1024, H=16, HD=64.
// - All GEMMs: v_wmma_f32_16x16x32_f16, fp32 accumulation.
// - Intermediates (Q,K,V,G,VG,M) kept in f16 workspace -> global->LDS staging
//   of those tiles is a byte-exact copy done with GLOBAL_LOAD_ASYNC_TO_LDS_B128
//   (ASYNCcnt / s_wait_asynccnt path).
// - f32 sources (x, weights, A) staged with packed v_cvt_pk_rtz_f16_f32.
// ---------------------------------------------------------------------------

#define BB   2
#define NN   2048
#define LL   1024
#define DD   1024
#define HH   16
#define HDIM 64
#define RR   (BB * NN)          // 4096 rows
#define EPSF 1e-5f

typedef __attribute__((ext_vector_type(16))) _Float16 v16h;
typedef __attribute__((ext_vector_type(8)))  _Float16 v8h;
typedef __attribute__((ext_vector_type(8)))  float    v8f;

// ---- WMMA helpers ----------------------------------------------------------

__device__ __forceinline__ v8f wmma_f16(v16h a, v16h b, v8f c) {
    // (neg_a, A, neg_b, B, c_mod, C, reuse_a, reuse_b)
    return __builtin_amdgcn_wmma_f32_16x16x32_f16(false, a, false, b,
                                                  (short)0, c, false, false);
}

__device__ __forceinline__ v16h frag16(const _Float16* p0, const _Float16* p1) {
    v8h lo = *(const v8h*)p0;
    v8h hi = *(const v8h*)p1;
    v16h r;
#pragma unroll
    for (int i = 0; i < 8; ++i) { r[i] = lo[i]; r[i + 8] = hi[i]; }
    return r;
}

// A-matrix 16x32 f16 fragment: lane (m = lane&15, h = lane>>4) holds
// K in two contiguous runs: [h*8, h*8+8) and [16+h*8, 16+h*8+8).
__device__ __forceinline__ v16h frag_a(const _Float16* rowp, int h) {
    return frag16(rowp + h * 8, rowp + 16 + h * 8);
}

// B-matrix 32x16 f16 fragment: lane (n = lane&15, h = lane>>4) holds
// K in two contiguous runs: [h*8, h*8+8) and [16+h*8, 16+h*8+8).
__device__ __forceinline__ v16h frag_b(const _Float16* colp, int h) {
    return frag16(colp + h * 8, colp + 16 + h * 8);
}

// ---- CDNA5 async global->LDS copy (byte-exact, 16B per lane) ---------------

__device__ __forceinline__ void async_copy_b128(_Float16* lds_dst, const _Float16* gsrc) {
    unsigned ldsoff = (unsigned)(uintptr_t)lds_dst;          // LDS byte offset
    unsigned long long gaddr = (unsigned long long)(uintptr_t)gsrc;
    asm volatile("global_load_async_to_lds_b128 %0, %1, off"
                 :: "v"(ldsoff), "v"(gaddr) : "memory");
}

__device__ __forceinline__ void wait_async0() {
    asm volatile("s_wait_asynccnt 0x0" ::: "memory");
}

// ---- packed f32->f16 store into LDS (v_cvt_pk_rtz_f16_f32 + b32 store) -----

__device__ __forceinline__ void store_pk(_Float16* dst, float a, float b) {
    auto p = __builtin_amdgcn_cvt_pkrtz(a, b);
    __builtin_memcpy((void*)dst, &p, 4);
}

// ---- reductions ------------------------------------------------------------

__device__ __forceinline__ float block_sum256(float v, float* red) {
#pragma unroll
    for (int o = 16; o > 0; o >>= 1) v += __shfl_xor(v, o, 32);
    int lane = threadIdx.x & 31, w = threadIdx.x >> 5;
    if (lane == 0) red[w] = v;
    __syncthreads();
    float t = 0.f;
#pragma unroll
    for (int i = 0; i < 8; ++i) t += red[i];
    __syncthreads();
    return t;
}

__device__ __forceinline__ float block_max256(float v, float* red) {
#pragma unroll
    for (int o = 16; o > 0; o >>= 1) v = fmaxf(v, __shfl_xor(v, o, 32));
    int lane = threadIdx.x & 31, w = threadIdx.x >> 5;
    if (lane == 0) red[w] = v;
    __syncthreads();
    float t = -3.402823466e38f;
#pragma unroll
    for (int i = 0; i < 8; ++i) t = fmaxf(t, red[i]);
    __syncthreads();
    return t;
}

// ---------------------------------------------------------------------------
// Kernel 1: Out = (X @ W + bias) * alpha.   X:(4096,1024) W:(1024,1024) f32.
// TX = float  -> packed-convert staging of X;  TX = _Float16 -> async staging.
// mode 0: f32 row-major (4096,1024)
// mode 1: f16 head-major   [z=b*H+h][n][hd]        (Q, K)
// mode 2: f16 head-transp. [z=b*H+h][hd][n]        (V, G)
// ---------------------------------------------------------------------------
template <typename TX>
__global__ __launch_bounds__(256)
void gemm_xw_kernel(const TX* __restrict__ X, const float* __restrict__ W,
                    const float* __restrict__ bias, void* __restrict__ OutV,
                    float alpha, int mode) {
    const int r0 = blockIdx.x * 64;
    const int c0 = blockIdx.y * 128;
    const int tid = threadIdx.x;
    const int lane = tid & 31, w = tid >> 5;
    const int mloc = lane & 15, hlf = lane >> 4;
    const int mrow = (w & 3) * 16;        // 0..48
    const int ncol = (w >> 2) * 64;       // 0 or 64

    __shared__ __align__(32) _Float16 sA[64 * 32];   // [m][k]
    __shared__ __align__(32) _Float16 sB[128 * 32];  // [n][k]

    v8f acc[4];
#pragma unroll
    for (int t = 0; t < 4; ++t)
#pragma unroll
        for (int i = 0; i < 8; ++i) acc[t][i] = 0.f;

    for (int kb = 0; kb < LL / 32; ++kb) {
        // ---- stage A tile 64x32 ----
        const int arow = tid >> 2, akq = (tid & 3) * 8;
        if constexpr (sizeof(TX) == 2) {
            // f16 source: byte-exact async copy, 16B per lane
            async_copy_b128(sA + arow * 32 + akq,
                            (const _Float16*)X + (size_t)(r0 + arow) * LL + kb * 32 + akq);
        } else {
            const float* gp = (const float*)X + (size_t)(r0 + arow) * LL + kb * 32 + akq;
            _Float16* sp = sA + arow * 32 + akq;
#pragma unroll
            for (int i = 0; i < 4; ++i) store_pk(sp + 2 * i, gp[2 * i], gp[2 * i + 1]);
            if (kb + 1 < LL / 32)
                __builtin_prefetch(gp + 32, 0, 1);
        }
        // ---- stage B tile 32x128 transposed to [n][k], packed converts ----
#pragma unroll
        for (int j = 0; j < 8; ++j) {
            int idx = tid + j * 256;              // 0..2047 pairs
            int n = idx & 127, k2 = (idx >> 7) * 2;
            float a0 = W[(size_t)(kb * 32 + k2) * DD + c0 + n];
            float a1 = W[(size_t)(kb * 32 + k2 + 1) * DD + c0 + n];
            store_pk(&sB[n * 32 + k2], a0, a1);
        }
        if (kb + 1 < LL / 32)
            __builtin_prefetch(W + (size_t)((kb + 1) * 32 + (tid & 31)) * DD + c0, 0, 1);

        if constexpr (sizeof(TX) == 2) wait_async0();
        __syncthreads();

        v16h a = frag_a(&sA[(mrow + mloc) * 32], hlf);
#pragma unroll
        for (int t = 0; t < 4; ++t) {
            v16h b = frag_b(&sB[(ncol + t * 16 + mloc) * 32], hlf);
            acc[t] = wmma_f16(a, b, acc[t]);
        }
        __syncthreads();
    }

    // epilogue: C/D layout -> lane N = lane&15; VGPR r -> M = r + 8*(lane>>4)
    float* Outf = (float*)OutV;
    _Float16* Outh = (_Float16*)OutV;
#pragma unroll
    for (int t = 0; t < 4; ++t) {
#pragma unroll
        for (int r = 0; r < 8; ++r) {
            int rg  = r0 + mrow + r + 8 * hlf;
            int col = c0 + ncol + t * 16 + mloc;
            float val = (acc[t][r] + bias[col]) * alpha;
            if (mode == 0) {
                Outf[(size_t)rg * DD + col] = val;
            } else {
                int b = rg >> 11, n = rg & 2047;
                int z = b * HH + (col >> 6), hd = col & 63;
                if (mode == 1)
                    Outh[((size_t)z * NN + n) * HDIM + hd] = (_Float16)val;
                else
                    Outh[((size_t)z * HDIM + hd) * NN + n] = (_Float16)val;
            }
        }
    }
}

// ---------------------------------------------------------------------------
// Kernel 2: V *= sigmoid(G)   (f16 head-transposed buffers, 4M elements)
// ---------------------------------------------------------------------------
__global__ __launch_bounds__(256)
void vg_kernel(_Float16* __restrict__ V, const _Float16* __restrict__ G) {
    size_t base = (size_t)blockIdx.x * 1024 + threadIdx.x;
#pragma unroll
    for (int i = 0; i < 4; ++i) {
        size_t idx = base + (size_t)i * 256;
        float g = (float)G[idx];
        float v = (float)V[idx];
        V[idx] = (_Float16)(v * (1.f / (1.f + __expf(-g))));
    }
}

// ---------------------------------------------------------------------------
// Kernel 3: raw scores S = Q @ K^T  (1/sqrt(64) folded into Q).
// Per (b,h): Q (2048x64) x K^T (64x2048). Block tile 64(M) x 256(N).
// Q/K are f16 [z][n][hd]: tiles staged with async global->LDS copies.
// ---------------------------------------------------------------------------
__global__ __launch_bounds__(256)
void scores_kernel(const _Float16* __restrict__ Qb, const _Float16* __restrict__ Kb,
                   float* __restrict__ Aout) {
    const int z  = blockIdx.z;                 // head index b*H+h
    const int m0 = blockIdx.x * 64;
    const int n0 = blockIdx.y * 256;
    const int tid = threadIdx.x;
    const int lane = tid & 31, w = tid >> 5;
    const int mloc = lane & 15, hlf = lane >> 4;
    const int mrow = (w & 3) * 16;
    const int ncol = (w >> 2) * 128;

    const _Float16* Qh = Qb + (size_t)z * NN * HDIM;
    const _Float16* Kh = Kb + (size_t)z * NN * HDIM;
    float* Ah = Aout + (size_t)z * NN * NN;

    __shared__ __align__(32) _Float16 sQ[64 * 64];    // [m][k]
    __shared__ __align__(32) _Float16 sK[256 * 64];   // [n][k]

    // async stage: byte-exact copies, 16B (8 f16) per lane per issue
#pragma unroll
    for (int j = 0; j < 2; ++j) {
        int c = tid + j * 256;                        // 512 chunks
        int row = c >> 3, kq = (c & 7) * 8;
        async_copy_b128(sQ + row * 64 + kq, Qh + (size_t)(m0 + row) * HDIM + kq);
    }
#pragma unroll
    for (int j = 0; j < 8; ++j) {
        int c = tid + j * 256;                        // 2048 chunks
        int row = c >> 3, kq = (c & 7) * 8;
        async_copy_b128(sK + row * 64 + kq, Kh + (size_t)(n0 + row) * HDIM + kq);
    }
    wait_async0();
    __syncthreads();

    v8f acc[8];
#pragma unroll
    for (int t = 0; t < 8; ++t)
#pragma unroll
        for (int i = 0; i < 8; ++i) acc[t][i] = 0.f;

#pragma unroll
    for (int kk = 0; kk < 2; ++kk) {
        v16h a = frag_a(&sQ[(mrow + mloc) * 64 + kk * 32], hlf);
#pragma unroll
        for (int t = 0; t < 8; ++t) {
            v16h b = frag_b(&sK[(ncol + t * 16 + mloc) * 64 + kk * 32], hlf);
            acc[t] = wmma_f16(a, b, acc[t]);
        }
    }

#pragma unroll
    for (int t = 0; t < 8; ++t)
#pragma unroll
        for (int r = 0; r < 8; ++r) {
            int M = m0 + mrow + r + 8 * hlf;
            int C = n0 + ncol + t * 16 + mloc;
            Ah[(size_t)M * NN + C] = acc[t][r];
        }
}

// ---------------------------------------------------------------------------
// Kernel 4: row softmax over A, in place. One block per row of 2048.
// ---------------------------------------------------------------------------
__global__ __launch_bounds__(256)
void softmax_kernel(float* __restrict__ A) {
    float* p = A + (size_t)blockIdx.x * NN;
    const int tid = threadIdx.x;
    __shared__ float red[8];

    float v[8];
#pragma unroll
    for (int i = 0; i < 8; ++i) v[i] = p[tid + i * 256];

    float mx = v[0];
#pragma unroll
    for (int i = 1; i < 8; ++i) mx = fmaxf(mx, v[i]);
    mx = block_max256(mx, red);

    float s = 0.f;
#pragma unroll
    for (int i = 0; i < 8; ++i) { v[i] = __expf(v[i] - mx); s += v[i]; }
    s = block_sum256(s, red);
    float inv = 1.f / s;

#pragma unroll
    for (int i = 0; i < 8; ++i) p[tid + i * 256] = v[i] * inv;
}

// ---------------------------------------------------------------------------
// Kernel 5: M = A @ (V*G) per head. A:(2048x2048) f32, VGt:[z][hd][key] f16.
// Block tile 128(M) x 64(N); K loop over 2048 in steps of 32.
// A staged with packed converts; VGt staged with async copies.
// Output M written f16 row-major (b, n, h*64+hd).
// ---------------------------------------------------------------------------
__global__ __launch_bounds__(256)
void av_kernel(const float* __restrict__ A, const _Float16* __restrict__ VGt,
               _Float16* __restrict__ Mout) {
    const int z  = blockIdx.z;
    const int m0 = blockIdx.x * 128;
    const int tid = threadIdx.x;
    const int lane = tid & 31, w = tid >> 5;
    const int mloc = lane & 15, hlf = lane >> 4;
    const int mrow = w * 16;

    const float* Ah = A + (size_t)z * NN * NN;
    const _Float16* Vh = VGt + (size_t)z * HDIM * NN;   // [hd][key]

    __shared__ __align__(32) _Float16 sA2[128 * 32];  // [m][k]
    __shared__ __align__(32) _Float16 sV[64 * 32];    // [hd][k]

    v8f acc[4];
#pragma unroll
    for (int t = 0; t < 4; ++t)
#pragma unroll
        for (int i = 0; i < 8; ++i) acc[t][i] = 0.f;

    for (int kb = 0; kb < NN / 32; ++kb) {
        // VG tile: byte-exact async copy (one 16B chunk per lane)
        {
            int hd = tid >> 2, kq = (tid & 3) * 8;
            async_copy_b128(sV + hd * 32 + kq,
                            Vh + (size_t)hd * NN + kb * 32 + kq);
        }
        // A tile 128x32 f32 -> f16 with packed converts (8 elems per chunk)
#pragma unroll
        for (int j = 0; j < 2; ++j) {
            int c = tid + j * 256;                    // 512 chunks
            int row = c >> 2, kq = (c & 3) * 8;
            const float* gp = Ah + (size_t)(m0 + row) * NN + kb * 32 + kq;
            _Float16* sp = sA2 + row * 32 + kq;
#pragma unroll
            for (int i = 0; i < 4; ++i) store_pk(sp + 2 * i, gp[2 * i], gp[2 * i + 1]);
        }
        if (kb + 1 < NN / 32)
            __builtin_prefetch(Ah + (size_t)(m0 + (tid >> 1)) * NN + (kb + 1) * 32, 0, 1);

        wait_async0();
        __syncthreads();

        v16h a = frag_a(&sA2[(mrow + mloc) * 32], hlf);
#pragma unroll
        for (int t = 0; t < 4; ++t) {
            v16h b = frag_b(&sV[(t * 16 + mloc) * 32], hlf);
            acc[t] = wmma_f16(a, b, acc[t]);
        }
        __syncthreads();
    }

    const int bidx = z >> 4, h = z & 15;
#pragma unroll
    for (int t = 0; t < 4; ++t)
#pragma unroll
        for (int r = 0; r < 8; ++r) {
            int M = m0 + mrow + r + 8 * hlf;
            int c = t * 16 + mloc;
            Mout[((size_t)bidx * NN + M) * DD + h * HDIM + c] = (_Float16)acc[t][r];
        }
}

// ---------------------------------------------------------------------------
// Kernel 6: y = x + P; out = gamma * (y - mu) * rsqrt(var + eps) + beta.
// One block per row of 1024.
// ---------------------------------------------------------------------------
__global__ __launch_bounds__(256)
void ln_kernel(const float* __restrict__ x, const float* __restrict__ P,
               const float* __restrict__ gamma, const float* __restrict__ beta,
               float* __restrict__ out) {
    const size_t base = (size_t)blockIdx.x * LL;
    const int tid = threadIdx.x;
    __shared__ float red[8];

    float y[4];
    float s = 0.f;
#pragma unroll
    for (int i = 0; i < 4; ++i) {
        int idx = tid + i * 256;
        y[i] = x[base + idx] + P[base + idx];
        s += y[i];
    }
    float mu = block_sum256(s, red) * (1.f / (float)LL);

    float q = 0.f;
#pragma unroll
    for (int i = 0; i < 4; ++i) { float d = y[i] - mu; q += d * d; }
    float var = block_sum256(q, red) * (1.f / (float)LL);
    float rs = rsqrtf(var + EPSF);

#pragma unroll
    for (int i = 0; i < 4; ++i) {
        int idx = tid + i * 256;
        out[base + idx] = gamma[idx] * (y[i] - mu) * rs + beta[idx];
    }
}

// ---------------------------------------------------------------------------
// Host-side launcher
// ---------------------------------------------------------------------------
extern "C" void kernel_launch(void* const* d_in, const int* in_sizes, int n_in,
                              void* d_out, int out_size, void* d_ws, size_t ws_size,
                              hipStream_t stream) {
    const float* x     = (const float*)d_in[0];
    const float* Wq    = (const float*)d_in[1];
    const float* bq    = (const float*)d_in[2];
    const float* Wk    = (const float*)d_in[3];
    const float* bk    = (const float*)d_in[4];
    const float* Wv    = (const float*)d_in[5];
    const float* bv    = (const float*)d_in[6];
    const float* Wg    = (const float*)d_in[7];
    const float* bg    = (const float*)d_in[8];
    const float* Wp    = (const float*)d_in[9];
    const float* bp    = (const float*)d_in[10];
    const float* gamma = (const float*)d_in[11];
    const float* beta  = (const float*)d_in[12];

    const size_t SZ = (size_t)RR * DD;                // 4M elements per buffer
    _Float16* hbase = (_Float16*)d_ws;
    _Float16* Qb = hbase + 0 * SZ;                    // f16 [z][n][hd]
    _Float16* Kb = hbase + 1 * SZ;                    // f16 [z][n][hd]
    _Float16* Vb = hbase + 2 * SZ;                    // f16 [z][hd][n] -> V*sig(G)
    _Float16* Gb = hbase + 3 * SZ;                    // f16 [z][hd][n]; reused as M
    _Float16* Mb = Gb;                                // f16 row-major (4096,1024)
    float*    Pb = (float*)(hbase + 4 * SZ);          // f32 row-major (4096,1024)

    float* out  = (float*)d_out;                      // (B,N,D)
    float* Aout = out + (size_t)RR * DD;              // (B,H,N,N)

    const dim3 blk(256);
    const dim3 ggrid(RR / 64, DD / 128);
    const float scale = 0.125f;                       // 1/sqrt(64), folded into Q

    // projections
    gemm_xw_kernel<float><<<ggrid, blk, 0, stream>>>(x, Wq, bq, Qb, scale, 1);
    gemm_xw_kernel<float><<<ggrid, blk, 0, stream>>>(x, Wk, bk, Kb, 1.f, 1);
    gemm_xw_kernel<float><<<ggrid, blk, 0, stream>>>(x, Wv, bv, Vb, 1.f, 2);
    gemm_xw_kernel<float><<<ggrid, blk, 0, stream>>>(x, Wg, bg, Gb, 1.f, 2);

    // V *= sigmoid(G)
    vg_kernel<<<dim3(RR * DD / 1024), blk, 0, stream>>>(Vb, Gb);

    // scores -> A (raw), then softmax in place
    scores_kernel<<<dim3(NN / 64, NN / 256, BB * HH), blk, 0, stream>>>(Qb, Kb, Aout);
    softmax_kernel<<<dim3(BB * HH * NN), blk, 0, stream>>>(Aout);

    // M = A @ (V*G)
    av_kernel<<<dim3(NN / 128, 1, BB * HH), blk, 0, stream>>>(Aout, Vb, Mb);

    // output projection (f16 X, async staging) + residual LayerNorm
    gemm_xw_kernel<_Float16><<<ggrid, blk, 0, stream>>>(Mb, Wp, bp, Pb, 1.f, 0);
    ln_kernel<<<dim3(RR), blk, 0, stream>>>(x, Pb, gamma, beta, out);
}
